// EquivariantBlock_54296976556717
// MI455X (gfx1250) — matrix operand order repaired and checked by reading, not compiled, not run on hardware
//
#include <hip/hip_runtime.h>
#include <hip/hip_bf16.h>

typedef __attribute__((ext_vector_type(16))) _Float16 v16h;
typedef __attribute__((ext_vector_type(8)))  float    v8f;

#define D_DIM 128

__device__ __forceinline__ float silu_f(float x)    { return x / (1.f + __expf(-x)); }
__device__ __forceinline__ float sigmoid_f(float x) { return 1.f / (1.f + __expf(-x)); }

__device__ __forceinline__ v8f wmma_f16(v16h a, v16h b, v8f c) {
  return __builtin_amdgcn_wmma_f32_16x16x32_f16(false, a, false, b, (short)0, c, false, false);
}

// A-fragment (16x32 f16) from a 128-float row in global memory.
// Lane layout per ISA: lanes 0-15 hold K={0..7,16..23}, lanes 16-31 K={8..15,24..31}.
__device__ __forceinline__ v16h fragA_row_f32(const float* __restrict__ r, int t, int koff) {
  const float* p0 = r + 32 * t + koff;
  v16h f;
#pragma unroll
  for (int i = 0; i < 8; ++i) { f[i] = (_Float16)p0[i]; f[8 + i] = (_Float16)p0[16 + i]; }
  return f;
}

__device__ __forceinline__ v16h fragA_diff_f32(const float* __restrict__ a,
                                               const float* __restrict__ b, int t, int koff) {
  const float* p0 = a + 32 * t + koff;
  const float* q0 = b + 32 * t + koff;
  v16h f;
#pragma unroll
  for (int i = 0; i < 8; ++i) {
    f[i]     = (_Float16)(p0[i]      - q0[i]);
    f[8 + i] = (_Float16)(p0[16 + i] - q0[16 + i]);
  }
  return f;
}

__device__ __forceinline__ v16h fragA_lds(const _Float16* __restrict__ r, int t, int koff) {
  const _Float16* p0 = r + 32 * t + koff;
  v16h f;
#pragma unroll
  for (int i = 0; i < 8; ++i) { f[i] = p0[i]; f[8 + i] = p0[16 + i]; }
  return f;
}

// K-tile covering the 3 concatenated scalar edge features (zero padded).
__device__ __forceinline__ v16h fragA_e3(float dist, float attr, float geo, int lane) {
  v16h f;
#pragma unroll
  for (int i = 0; i < 16; ++i) f[i] = (_Float16)0.f;
  if (lane < 16) { f[0] = (_Float16)dist; f[1] = (_Float16)attr; f[2] = (_Float16)geo; }
  return f;
}

// B fragment: pre-swizzled f16 weights, one contiguous 32B load per lane.
__device__ __forceinline__ v16h fragB_g(const _Float16* __restrict__ W, int t, int n, int lane) {
  return *reinterpret_cast<const v16h*>(W + (size_t)(((t * 8 + n) * 32 + lane) * 16));
}

__device__ __forceinline__ float reduce16(float v) {
  v += __shfl_xor(v, 1, 32);
  v += __shfl_xor(v, 2, 32);
  v += __shfl_xor(v, 4, 32);
  v += __shfl_xor(v, 8, 32);
  return v;
}

// ---------------- weight prep: f32 [K,128] row-major -> f16 B-fragment order ---------------
__global__ void prep_weight_kernel(const float* __restrict__ W, _Float16* __restrict__ Wf,
                                   int Kact, int total) {
  int idx = blockIdx.x * blockDim.x + threadIdx.x;
  if (idx >= total) return;
  int j    = idx & 15;
  int lane = (idx >> 4) & 31;
  int n    = (idx >> 9) & 7;
  int t    = idx >> 12;
  int K    = 32 * t + 8 * (lane >> 4) + (j & 7) + 16 * (j >> 3);
  int Ncol = 16 * n + (lane & 15);
  float v  = (K < Kact) ? W[(size_t)K * D_DIM + Ncol] : 0.f;
  Wf[idx]  = (_Float16)v;
}

__global__ void zero_f32_kernel(float* __restrict__ p, int n) {
  int i = blockIdx.x * blockDim.x + threadIdx.x;
  if (i < n) p[i] = 0.f;
}

// ---------------- hh = h @ W_lin + b_lin ----------------
__global__ void __launch_bounds__(128) node_lin_kernel(
    const float* __restrict__ h, const _Float16* __restrict__ Wf,
    const float* __restrict__ bias, float* __restrict__ hh, int N) {
  int lane  = threadIdx.x & 31;
  int wave  = threadIdx.x >> 5;
  int nbase = (blockIdx.x * 4 + wave) * 16;
  if (nbase >= N) return;
  int nlo = lane & 15, half = lane >> 4;
  int koff = half ? 8 : 0;
  const float* arow = h + (size_t)(nbase + nlo) * D_DIM;
  v8f acc[8];
#pragma unroll
  for (int n = 0; n < 8; ++n)
#pragma unroll
    for (int v = 0; v < 8; ++v) acc[n][v] = 0.f;
#pragma unroll
  for (int t = 0; t < 4; ++t) {
    v16h a = fragA_row_f32(arow, t, koff);
#pragma unroll
    for (int n = 0; n < 8; ++n) acc[n] = wmma_f16(a, fragB_g(Wf, t, n, lane), acc[n]);
  }
#pragma unroll
  for (int n = 0; n < 8; ++n) {
    float b = bias[16 * n + nlo];
#pragma unroll
    for (int v = 0; v < 8; ++v)
      hh[(size_t)(nbase + v + 8 * half) * D_DIM + 16 * n + nlo] = acc[n][v] + b;
  }
}

// ---------------- edge kernel 1: attention + message MLPs + segment-sum ----------------
__global__ void __launch_bounds__(128) edge_msg_kernel(
    const float* __restrict__ hh, const float* __restrict__ x,
    const int* __restrict__ erow, const int* __restrict__ ecol,
    const float* __restrict__ edge_attr, const float* __restrict__ edge_mask,
    const _Float16* __restrict__ Wa1f, const float* __restrict__ ba1,
    const float* __restrict__ Wa2,   const float* __restrict__ ba2,
    const _Float16* __restrict__ We1f, const float* __restrict__ be1,
    const _Float16* __restrict__ We2f, const float* __restrict__ be2,
    float* __restrict__ agg, float* __restrict__ distbuf, float* __restrict__ geobuf, int E) {
  __shared__ _Float16 hbuf[4][16 * D_DIM];
  int lane  = threadIdx.x & 31;
  int wave  = threadIdx.x >> 5;
  int ebase = (blockIdx.x * 4 + wave) * 16;
  int nlo = lane & 15, half = lane >> 4;
  int koff = half ? 8 : 0;
  bool active = ebase < E;

  v8f acc[8];
  float attv[8];
  _Float16* buf = hbuf[wave];

  if (active) {
    int e  = ebase + nlo;
    int ri = erow[e], ci = ecol[e];
    const float* hr = hh + (size_t)ri * D_DIM;
    const float* hc = hh + (size_t)ci * D_DIM;

    // geodesic feature ||hr - hc|| (halves cover complementary K, combine via xor-16)
    float g2 = 0.f;
#pragma unroll
    for (int t = 0; t < 4; ++t) {
      const float* p0 = hr + 32 * t + koff;
      const float* q0 = hc + 32 * t + koff;
#pragma unroll
      for (int i = 0; i < 8; ++i) {
        float d0 = p0[i] - q0[i];
        float d1 = p0[16 + i] - q0[16 + i];
        g2 += d0 * d0 + d1 * d1;
      }
    }
    g2 += __shfl_xor(g2, 16, 32);
    float geo = sqrtf(g2 + 1e-8f);

    float dx = x[ri * 3 + 0] - x[ci * 3 + 0];
    float dy = x[ri * 3 + 1] - x[ci * 3 + 1];
    float dz = x[ri * 3 + 2] - x[ci * 3 + 2];
    float dist = sqrtf(dx * dx + dy * dy + dz * dz + 1e-8f);
    float attr = edge_attr[e];
    if (lane < 16) { distbuf[e] = dist; geobuf[e] = geo; }

    // att = sigmoid(silu([hr,hc,e3] @ Wa1 + ba1) @ Wa2 + ba2) * edge_mask
#pragma unroll
    for (int n = 0; n < 8; ++n)
#pragma unroll
      for (int v = 0; v < 8; ++v) acc[n][v] = 0.f;
#pragma unroll
    for (int t = 0; t < 9; ++t) {
      v16h a = (t < 4) ? fragA_row_f32(hr, t, koff)
             : (t < 8) ? fragA_row_f32(hc, t - 4, koff)
                       : fragA_e3(dist, attr, geo, lane);
#pragma unroll
      for (int n = 0; n < 8; ++n) acc[n] = wmma_f16(a, fragB_g(Wa1f, t, n, lane), acc[n]);
    }
    float ba1l[8], wa2l[8];
#pragma unroll
    for (int n = 0; n < 8; ++n) { ba1l[n] = ba1[16 * n + nlo]; wa2l[n] = Wa2[16 * n + nlo]; }
    float ba2v = ba2[0];
#pragma unroll
    for (int v = 0; v < 8; ++v) {
      float p = 0.f;
#pragma unroll
      for (int n = 0; n < 8; ++n) p += silu_f(acc[n][v] + ba1l[n]) * wa2l[n];
      p = reduce16(p);
      attv[v] = sigmoid_f(p + ba2v) * edge_mask[ebase + v + 8 * half];
    }

    // msg layer 1: silu([hc-hr, e3] @ We1 + be1)
#pragma unroll
    for (int n = 0; n < 8; ++n)
#pragma unroll
      for (int v = 0; v < 8; ++v) acc[n][v] = 0.f;
#pragma unroll
    for (int t = 0; t < 5; ++t) {
      v16h a = (t < 4) ? fragA_diff_f32(hc, hr, t, koff)
                       : fragA_e3(dist, attr, geo, lane);
#pragma unroll
      for (int n = 0; n < 8; ++n) acc[n] = wmma_f16(a, fragB_g(We1f, t, n, lane), acc[n]);
    }
#pragma unroll
    for (int n = 0; n < 8; ++n) {
      float b = be1[16 * n + nlo];
#pragma unroll
      for (int v = 0; v < 8; ++v)
        buf[(v + 8 * half) * D_DIM + 16 * n + nlo] = (_Float16)silu_f(acc[n][v] + b);
    }
  }
  __syncthreads();
  if (active) {
    // msg layer 2 + attention gate + atomic segment-sum into agg[row]
    const _Float16* lrow = buf + nlo * D_DIM;
#pragma unroll
    for (int n = 0; n < 8; ++n)
#pragma unroll
      for (int v = 0; v < 8; ++v) acc[n][v] = 0.f;
#pragma unroll
    for (int t = 0; t < 4; ++t) {
      v16h a = fragA_lds(lrow, t, koff);
#pragma unroll
      for (int n = 0; n < 8; ++n) acc[n] = wmma_f16(a, fragB_g(We2f, t, n, lane), acc[n]);
    }
    int rIdx[8];
#pragma unroll
    for (int v = 0; v < 8; ++v) rIdx[v] = erow[ebase + v + 8 * half];
#pragma unroll
    for (int n = 0; n < 8; ++n) {
      float b = be2[16 * n + nlo];
#pragma unroll
      for (int v = 0; v < 8; ++v) {
        float val = (acc[n][v] + b) * attv[v];
        atomicAdd(&agg[(size_t)rIdx[v] * D_DIM + 16 * n + nlo], val);
      }
    }
  }
}

// ---------------- node kernel: out MLP + residual + LayerNorm + silu ----------------
__global__ void __launch_bounds__(64) node_out_kernel(
    const float* __restrict__ agg, const float* __restrict__ hh,
    const _Float16* __restrict__ Wn1f, const float* __restrict__ bn1,
    const _Float16* __restrict__ Wn2f, const float* __restrict__ bn2,
    const float* __restrict__ ln_g, const float* __restrict__ ln_b,
    float* __restrict__ hout, int N) {
  __shared__ _Float16 hbuf[2][16 * D_DIM];
  int lane  = threadIdx.x & 31;
  int wave  = threadIdx.x >> 5;
  int nbase = (blockIdx.x * 2 + wave) * 16;
  int nlo = lane & 15, half = lane >> 4;
  int koff = half ? 8 : 0;
  bool active = nbase < N;
  v8f acc[8];
  _Float16* buf = hbuf[wave];

  if (active) {
    const float* arow = agg + (size_t)(nbase + nlo) * D_DIM;
#pragma unroll
    for (int n = 0; n < 8; ++n)
#pragma unroll
      for (int v = 0; v < 8; ++v) acc[n][v] = 0.f;
#pragma unroll
    for (int t = 0; t < 4; ++t) {
      v16h a = fragA_row_f32(arow, t, koff);
#pragma unroll
      for (int n = 0; n < 8; ++n) acc[n] = wmma_f16(a, fragB_g(Wn1f, t, n, lane), acc[n]);
    }
#pragma unroll
    for (int n = 0; n < 8; ++n) {
      float b = bn1[16 * n + nlo];
#pragma unroll
      for (int v = 0; v < 8; ++v)
        buf[(v + 8 * half) * D_DIM + 16 * n + nlo] = (_Float16)silu_f(acc[n][v] + b);
    }
  }
  __syncthreads();
  if (active) {
    const _Float16* lrow = buf + nlo * D_DIM;
#pragma unroll
    for (int n = 0; n < 8; ++n)
#pragma unroll
      for (int v = 0; v < 8; ++v) acc[n][v] = 0.f;
#pragma unroll
    for (int t = 0; t < 4; ++t) {
      v16h a = fragA_lds(lrow, t, koff);
#pragma unroll
      for (int n = 0; n < 8; ++n) acc[n] = wmma_f16(a, fragB_g(Wn2f, t, n, lane), acc[n]);
    }
    float val[8][8];
#pragma unroll
    for (int n = 0; n < 8; ++n) {
      float b = bn2[16 * n + nlo];
#pragma unroll
      for (int v = 0; v < 8; ++v)
        val[n][v] = hh[(size_t)(nbase + v + 8 * half) * D_DIM + 16 * n + nlo] + acc[n][v] + b;
    }
#pragma unroll
    for (int v = 0; v < 8; ++v) {
      float s = 0.f;
#pragma unroll
      for (int n = 0; n < 8; ++n) s += val[n][v];
      float mu = reduce16(s) * (1.f / 128.f);
      float q = 0.f;
#pragma unroll
      for (int n = 0; n < 8; ++n) { float d = val[n][v] - mu; q += d * d; }
      float rstd = rsqrtf(reduce16(q) * (1.f / 128.f) + 1e-5f);
#pragma unroll
      for (int n = 0; n < 8; ++n) {
        int c = 16 * n + nlo;
        float o = (val[n][v] - mu) * rstd * ln_g[c] + ln_b[c];
        hout[(size_t)(nbase + v + 8 * half) * D_DIM + c] = silu_f(o);
      }
    }
  }
}

// ---------------- edge kernel 2: coord MLP + coordinate segment-sum ----------------
__global__ void __launch_bounds__(128) edge_coord_kernel(
    const float* __restrict__ hh2, const float* __restrict__ x,
    const int* __restrict__ erow, const int* __restrict__ ecol,
    const float* __restrict__ edge_attr, const float* __restrict__ edge_mask,
    const float* __restrict__ distbuf, const float* __restrict__ geobuf,
    const _Float16* __restrict__ Wc1f, const float* __restrict__ bc1,
    const _Float16* __restrict__ Wc2f, const float* __restrict__ bc2,
    const float* __restrict__ Wc3, float* __restrict__ aggx, int E) {
  __shared__ _Float16 hbuf[4][16 * D_DIM];
  int lane  = threadIdx.x & 31;
  int wave  = threadIdx.x >> 5;
  int ebase = (blockIdx.x * 4 + wave) * 16;
  int nlo = lane & 15, half = lane >> 4;
  int koff = half ? 8 : 0;
  bool active = ebase < E;
  v8f acc[8];
  _Float16* buf = hbuf[wave];

  if (active) {
    int e  = ebase + nlo;
    int ri = erow[e], ci = ecol[e];
    const float* hr = hh2 + (size_t)ri * D_DIM;
    const float* hc = hh2 + (size_t)ci * D_DIM;
    float dist = distbuf[e];
    float attr = edge_attr[e];
    float geo  = geobuf[e];
#pragma unroll
    for (int n = 0; n < 8; ++n)
#pragma unroll
      for (int v = 0; v < 8; ++v) acc[n][v] = 0.f;
#pragma unroll
    for (int t = 0; t < 9; ++t) {
      v16h a = (t < 4) ? fragA_row_f32(hr, t, koff)
             : (t < 8) ? fragA_row_f32(hc, t - 4, koff)
                       : fragA_e3(dist, attr, geo, lane);
#pragma unroll
      for (int n = 0; n < 8; ++n) acc[n] = wmma_f16(a, fragB_g(Wc1f, t, n, lane), acc[n]);
    }
#pragma unroll
    for (int n = 0; n < 8; ++n) {
      float b = bc1[16 * n + nlo];
#pragma unroll
      for (int v = 0; v < 8; ++v)
        buf[(v + 8 * half) * D_DIM + 16 * n + nlo] = (_Float16)silu_f(acc[n][v] + b);
    }
  }
  __syncthreads();
  if (active) {
    const _Float16* lrow = buf + nlo * D_DIM;
#pragma unroll
    for (int n = 0; n < 8; ++n)
#pragma unroll
      for (int v = 0; v < 8; ++v) acc[n][v] = 0.f;
#pragma unroll
    for (int t = 0; t < 4; ++t) {
      v16h a = fragA_lds(lrow, t, koff);
#pragma unroll
      for (int n = 0; n < 8; ++n) acc[n] = wmma_f16(a, fragB_g(Wc2f, t, n, lane), acc[n]);
    }
    float bc2l[8], wc3l[8];
#pragma unroll
    for (int n = 0; n < 8; ++n) { bc2l[n] = bc2[16 * n + nlo]; wc3l[n] = Wc3[16 * n + nlo]; }
    float mv[8];
#pragma unroll
    for (int v = 0; v < 8; ++v) {
      float p = 0.f;
#pragma unroll
      for (int n = 0; n < 8; ++n) p += silu_f(acc[n][v] + bc2l[n]) * wc3l[n];
      mv[v] = reduce16(p);
    }
    if (nlo == 0) {
#pragma unroll
      for (int v = 0; v < 8; ++v) {
        int ev = ebase + v + 8 * half;
        int rv = erow[ev], cv = ecol[ev];
        float s = mv[v] * edge_mask[ev] / (distbuf[ev] + 1.f);
#pragma unroll
        for (int k = 0; k < 3; ++k) {
          float dd = x[rv * 3 + k] - x[cv * 3 + k];
          atomicAdd(&aggx[rv * 3 + k], dd * s);
        }
      }
    }
  }
}

__global__ void coord_out_kernel(const float* __restrict__ x, const float* __restrict__ aggx,
                                 const float* __restrict__ node_mask, float* __restrict__ xout,
                                 int n3) {
  int i = blockIdx.x * blockDim.x + threadIdx.x;
  if (i >= n3) return;
  int node = i / 3;
  xout[i] = (x[i] + aggx[i] * 0.01f) * node_mask[node];
}

extern "C" void kernel_launch(void* const* d_in, const int* in_sizes, int n_in,
                              void* d_out, int out_size, void* d_ws, size_t ws_size,
                              hipStream_t stream) {
  (void)n_in; (void)out_size; (void)ws_size;
  const float* h         = (const float*)d_in[0];
  const float* x         = (const float*)d_in[1];
  const int*   ei        = (const int*)d_in[2];
  const float* node_mask = (const float*)d_in[3];
  const float* edge_mask = (const float*)d_in[4];
  const float* edge_attr = (const float*)d_in[5];
  const float* W_lin = (const float*)d_in[6];  const float* b_lin = (const float*)d_in[7];
  const float* We1   = (const float*)d_in[8];  const float* be1   = (const float*)d_in[9];
  const float* We2   = (const float*)d_in[10]; const float* be2   = (const float*)d_in[11];
  const float* Wn1   = (const float*)d_in[12]; const float* bn1   = (const float*)d_in[13];
  const float* Wn2   = (const float*)d_in[14]; const float* bn2   = (const float*)d_in[15];
  const float* Wa1   = (const float*)d_in[16]; const float* ba1   = (const float*)d_in[17];
  const float* Wa2   = (const float*)d_in[18]; const float* ba2   = (const float*)d_in[19];
  const float* ln_g  = (const float*)d_in[20]; const float* ln_b  = (const float*)d_in[21];
  const float* Wc1   = (const float*)d_in[22]; const float* bc1   = (const float*)d_in[23];
  const float* Wc2   = (const float*)d_in[24]; const float* bc2   = (const float*)d_in[25];
  const float* Wc3   = (const float*)d_in[26];

  const int N = in_sizes[0] / D_DIM;
  const int E = in_sizes[2] / 2;
  const int* erow = ei;
  const int* ecol = ei + E;

  char* ws = (char*)d_ws;
  size_t off = 0;
  auto take = [&](size_t bytes) -> char* {
    char* p = ws + off;
    off = (off + bytes + 255) & ~(size_t)255;
    return p;
  };
  float* agg      = (float*)take((size_t)N * D_DIM * 4);
  float* aggx     = (float*)take((size_t)N * 3 * 4);
  float* hhbuf    = (float*)take((size_t)N * D_DIM * 4);
  float* distb    = (float*)take((size_t)E * 4);
  float* geob     = (float*)take((size_t)E * 4);
  _Float16* Wlinf = (_Float16*)take((size_t)4 * 4096 * 2);
  _Float16* Wa1f  = (_Float16*)take((size_t)9 * 4096 * 2);
  _Float16* We1f  = (_Float16*)take((size_t)5 * 4096 * 2);
  _Float16* We2f  = (_Float16*)take((size_t)4 * 4096 * 2);
  _Float16* Wn1f  = (_Float16*)take((size_t)4 * 4096 * 2);
  _Float16* Wn2f  = (_Float16*)take((size_t)4 * 4096 * 2);
  _Float16* Wc1f  = (_Float16*)take((size_t)9 * 4096 * 2);
  _Float16* Wc2f  = (_Float16*)take((size_t)4 * 4096 * 2);

  auto prep = [&](const float* W, _Float16* Wf, int Kact, int Ktiles) {
    int total = Ktiles * 4096;
    prep_weight_kernel<<<(total + 255) / 256, 256, 0, stream>>>(W, Wf, Kact, total);
  };
  prep(W_lin, Wlinf, 128, 4);
  prep(Wa1,   Wa1f,  259, 9);
  prep(We1,   We1f,  131, 5);
  prep(We2,   We2f,  128, 4);
  prep(Wn1,   Wn1f,  128, 4);
  prep(Wn2,   Wn2f,  128, 4);
  prep(Wc1,   Wc1f,  259, 9);
  prep(Wc2,   Wc2f,  128, 4);

  zero_f32_kernel<<<(N * D_DIM + 255) / 256, 256, 0, stream>>>(agg, N * D_DIM);
  zero_f32_kernel<<<(N * 3 + 255) / 256, 256, 0, stream>>>(aggx, N * 3);

  const int ntiles = (N + 15) / 16;
  node_lin_kernel<<<(ntiles + 3) / 4, 128, 0, stream>>>(h, Wlinf, b_lin, hhbuf, N);

  const int etiles = (E + 15) / 16;
  edge_msg_kernel<<<(etiles + 3) / 4, 128, 0, stream>>>(
      hhbuf, x, erow, ecol, edge_attr, edge_mask,
      Wa1f, ba1, Wa2, ba2, We1f, be1, We2f, be2,
      agg, distb, geob, E);

  float* hout = (float*)d_out;
  float* xout = hout + (size_t)N * D_DIM;
  node_out_kernel<<<(ntiles + 1) / 2, 64, 0, stream>>>(
      agg, hhbuf, Wn1f, bn1, Wn2f, bn2, ln_g, ln_b, hout, N);

  edge_coord_kernel<<<(etiles + 3) / 4, 128, 0, stream>>>(
      hout, x, erow, ecol, edge_attr, edge_mask, distb, geob,
      Wc1f, bc1, Wc2f, bc2, Wc3, aggx, E);

  coord_out_kernel<<<(N * 3 + 255) / 256, 256, 0, stream>>>(x, aggx, node_mask, xout, N * 3);
}
